// GTPPO_model_22720376996175
// MI455X (gfx1250) — matrix operand order repaired and checked by reading, not compiled, not run on hardware
//
#include <hip/hip_runtime.h>
#include <hip/hip_bf16.h>
#include <math.h>

// Problem constants (from reference)
#define BB 8
#define PP 64
#define TT 19
#define HH 64
#define HG 32
#define SLOPE 0.2f

typedef float v2f __attribute__((ext_vector_type(2)));
typedef float v8f __attribute__((ext_vector_type(8)));
typedef _Float16 v16h __attribute__((ext_vector_type(16)));

#ifndef __has_builtin
#define __has_builtin(x) 0
#endif

#if __has_builtin(__builtin_amdgcn_wmma_f32_16x16x4_f32)
#define USE_F32_WMMA 1
#else
#define USE_F32_WMMA 0
#endif

// Fully fused GAT kernel. Block = (b, t, mi): mi selects a 16-row i-tile.
// 128 threads = 4 wave32 waves; each wave produces one 16x16 output tile
// via v_wmma_f32_16x16x4_f32 with fused sigmoid.
//
// Key algebra: s1 = (agents @ W1^T) @ w2[:HG] = agents @ (W1^T w2[:HG]) = agents . v1
// so the (B,P,T,HG) "Wa" intermediate never exists; v1/v2 are recomputed per
// block (W1 is 8KB -> L2 resident) to avoid a second kernel launch.
__global__ __launch_bounds__(128) void gtppo_fused(const float* __restrict__ agents,
                                                   const float* __restrict__ adj,
                                                   const unsigned char* __restrict__ mask,
                                                   const float* __restrict__ W1,
                                                   const float* __restrict__ w2,
                                                   float* __restrict__ out) {
    __shared__ float Ag[PP * 65];     // agents[b,:,t,:]  (64x64, stride 65: conflict-free columns)
    __shared__ float Wt[16 * 65];     // attention weights tile (16 rows i, 64 cols j)
    __shared__ float v1[HH], v2[HH];
    __shared__ float s1[16], s2[PP];
    __shared__ unsigned char msk[PP];

    const int blk = blockIdx.x;           // 0 .. B*T*4-1
    const int mi  = blk & 3;              // i-tile (rows mi*16 .. mi*16+15)
    const int bt  = blk >> 2;
    const int b   = bt / TT;
    const int t   = bt % TT;
    const int tid = threadIdx.x;

    // --- Issue adjacency-gate loads into registers NOW; consumed after the
    // softmax barrier, so their latency hides under staging/logits/softmax.
    // Index mapping must match the gate phase below.
    float greg[8];
#pragma unroll
    for (int q = 0; q < 8; ++q) {
        const int idx = tid + q * 128;          // 0..1023
        const int ii = idx >> 6, j = idx & 63;
        greg[q] = adj[((b * PP + mi * 16 + ii) * PP + j) * TT + t];
    }

    // --- Phase 1 (wave-uniform split): waves 0-1 build v1/v2 + mask,
    // waves 2-3 stage the agents tile with coalesced b128 loads.
    if (tid < HH) {
        float a1 = 0.f, a2 = 0.f;
        for (int g = 0; g < HG; ++g) {
            const float w = W1[g * HH + tid];   // W1 is (HG,H) row-major
            a1 += w * w2[g];
            a2 += w * w2[HG + g];
        }
        v1[tid] = a1;
        v2[tid] = a2;
        msk[tid] = mask[b * PP + tid];
    } else {
        const int t2 = tid - 64;
#pragma unroll
        for (int q = 0; q < 16; ++q) {
            const int idx = t2 + q * 64;        // float4 index 0..1023
            const int j  = idx >> 4;            // 16 float4 per 64-float row
            const int h4 = (idx & 15) << 2;
            const float4 val =
                *(const float4*)&agents[((b * PP + j) * TT + t) * HH + h4];
            float* dst = &Ag[j * 65 + h4];      // stride-65 row: keep scalar stores
            dst[0] = val.x; dst[1] = val.y; dst[2] = val.z; dst[3] = val.w;
        }
    }
    __syncthreads();

    // --- s2[j] for all 64 j; s1 for this block's 16 i-rows.
    // Column access: bank = (i + h) % 64 -> conflict-free with stride 65.
    if (tid < PP) {
        const float* row = &Ag[tid * 65];
        float a2 = 0.f;
        for (int h = 0; h < HH; ++h) a2 += row[h] * v2[h];
        s2[tid] = a2;
        if (tid < 16) {
            const float* rowi = &Ag[(mi * 16 + tid) * 65];
            float a1 = 0.f;
            for (int h = 0; h < HH; ++h) a1 += rowi[h] * v1[h];
            s1[tid] = a1;
        }
    }
    __syncthreads();

    // --- Logits: leaky_relu(s1_i + s2_j), then neighbor mask -> -1e9
    for (int idx = tid; idx < 16 * PP; idx += 128) {
        const int ii = idx >> 6, j = idx & 63;
        float x = s1[ii] + s2[j];
        x = (x > 0.f) ? x : SLOPE * x;
        if (msk[j]) x = -1e9f;
        Wt[ii * 65 + j] = x;
    }
    __syncthreads();

    // --- Row softmax over j (16 rows, one thread each; sequential-bank rows)
    if (tid < 16) {
        float* row = &Wt[tid * 65];
        float mx = -INFINITY;
        for (int j = 0; j < PP; ++j) mx = fmaxf(mx, row[j]);
        float sum = 0.f;
        for (int j = 0; j < PP; ++j) {
            const float e = __expf(row[j] - mx);
            row[j] = e;
            sum += e;
        }
        const float inv = 1.f / sum;
        for (int j = 0; j < PP; ++j) row[j] *= inv;
    }
    __syncthreads();

    // --- Hard gate from pre-loaded registers: zero where adj <= 0
#pragma unroll
    for (int q = 0; q < 8; ++q) {
        const int idx = tid + q * 128;
        const int ii = idx >> 6, j = idx & 63;
        if (!(greg[q] > 0.f)) Wt[ii * 65 + j] = 0.f;
    }
    __syncthreads();

    // --- WMMA: out_tile(16x16) = Wt(16x64) @ Ag(64x, 16-col slice).
    // All 128 threads reach here (EXEC all-ones as WMMA requires).
    const int wv = tid >> 5;          // wave -> h-tile
    const int l  = tid & 31;
    const int m  = l & 15;            // M (A rows) / N (B cols)
    const int ch = l >> 4;            // lane-half selects K pair
    const int hbase = wv * 16;

    v8f acc = {0.f, 0.f, 0.f, 0.f, 0.f, 0.f, 0.f, 0.f};

#if USE_F32_WMMA
    // V_WMMA_F32_16X16X4_F32: A lane l -> M=l%16, K = k0 + 2*(l/16) + {0,1};
    //                         B lane l -> N=l%16, same K striping.
    for (int k0 = 0; k0 < PP; k0 += 4) {
        const int ka = k0 + 2 * ch;
        v2f a, bm;
        a.x  = Wt[m * 65 + ka];
        a.y  = Wt[m * 65 + ka + 1];
        bm.x = Ag[ka * 65 + hbase + m];
        bm.y = Ag[(ka + 1) * 65 + hbase + m];
        acc = __builtin_amdgcn_wmma_f32_16x16x4_f32(
            /*neg_a=*/false, a, /*neg_b=*/false, bm,
            /*c_mod=*/(short)0, acc, /*reuse_a=*/false, /*reuse_b=*/false);
    }
#else
    // Fallback (codegen-confirmed builtin): f16 in, f32 accumulate, K=32.
    for (int k0 = 0; k0 < PP; k0 += 32) {
        v16h a, bm;
        for (int e = 0; e < 16; ++e) {
            const int ka = k0 + ((e < 8) ? (8 * ch + e) : (16 + 8 * ch + (e - 8)));
            a[e] = (_Float16)Wt[m * 65 + ka];
            const int kb = k0 + ((e < 8) ? (8 * ch + e) : (16 + 8 * ch + (e - 8)));
            bm[e] = (_Float16)Ag[kb * 65 + hbase + m];
        }
        acc = __builtin_amdgcn_wmma_f32_16x16x32_f16(
            false, a, false, bm, (short)0, acc, false, false);
    }
#endif

    // --- C/D layout: VGPR r -> row M = r + 8*(l/16), col N = l%16. Fused sigmoid.
#pragma unroll
    for (int r = 0; r < 8; ++r) {
        const int i = mi * 16 + r + 8 * ch;
        const float y = 1.f / (1.f + __expf(-acc[r]));
        out[((b * PP + i) * TT + t) * HH + hbase + m] = y;
    }
}

extern "C" void kernel_launch(void* const* d_in, const int* in_sizes, int n_in,
                              void* d_out, int out_size, void* d_ws, size_t ws_size,
                              hipStream_t stream) {
    const float*         agents = (const float*)d_in[0];          // (B,P,T,H) f32
    const float*         adj    = (const float*)d_in[1];          // (B,P,P,T) f32
    const unsigned char* mask   = (const unsigned char*)d_in[2];  // (B,P) bool (1B)
    const float*         W1     = (const float*)d_in[3];          // (HG,H) f32
    const float*         w2     = (const float*)d_in[4];          // (2*HG,) f32
    float* out = (float*)d_out;                                   // (B,P,T,H) f32
    (void)d_ws; (void)ws_size; (void)in_sizes; (void)n_in; (void)out_size;

    gtppo_fused<<<BB * TT * 4, 128, 0, stream>>>(agents, adj, mask, W1, w2, out);
}
